// CausalSelfAttention_30691836297463
// MI455X (gfx1250) — matrix operand order repaired and checked by reading, not compiled, not run on hardware
//
#include <hip/hip_runtime.h>

typedef __attribute__((ext_vector_type(16))) __bf16 v16bf;
typedef __attribute__((ext_vector_type(8)))  __bf16 v8bf;
typedef __attribute__((ext_vector_type(8)))  float  v8f;

constexpr int cB = 2, cT = 2048, cC = 1024, cH = 16, cD = 64;

#define DEVINL __device__ __forceinline__

// A-operand (16x32 bf16): lane = M, element i <-> K = (i%8) + 16*(i/8) + 8*half
DEVINL v16bf load_a_bf16(const __bf16* row, int k0, int hv) {
  v8bf lo = *(const v8bf*)(row + k0 + hv * 8);
  v8bf hi = *(const v8bf*)(row + k0 + 16 + hv * 8);
  v16bf a;
#pragma unroll
  for (int i = 0; i < 8; ++i) { a[i] = lo[i]; a[i + 8] = hi[i]; }
  return a;
}

DEVINL v16bf load_a_f32(const float* row, int k0, int hv) {
  const float* p0 = row + k0 + hv * 8;
  const float* p1 = row + k0 + 16 + hv * 8;
  v16bf a;
#pragma unroll
  for (int i = 0; i < 8; ++i) { a[i] = (__bf16)p0[i]; a[i + 8] = (__bf16)p1[i]; }
  return a;
}

// B-operand (32x16 bf16): lane = N, element i <-> K = (i%8) + 16*(i/8) + 8*half
DEVINL v16bf load_b_bf16(const __bf16* row, int k0, int hv) {
  v8bf lo = *(const v8bf*)(row + k0 + hv * 8);
  v8bf hi = *(const v8bf*)(row + k0 + 16 + hv * 8);
  v16bf b;
#pragma unroll
  for (int i = 0; i < 8; ++i) { b[i] = lo[i]; b[i + 8] = hi[i]; }
  return b;
}

DEVINL v8f wmma_bf16(v16bf a, v16bf b, v8f c) {
  return __builtin_amdgcn_wmma_f32_16x16x32_bf16(false, a, false, b, (short)0, c,
                                                 false, false);
}

DEVINL v8f zero_v8f() {
  v8f z = {0.f, 0.f, 0.f, 0.f, 0.f, 0.f, 0.f, 0.f};
  return z;
}

// ---------------------------------------------------------------------------
// fp32 [K][N] row-major  ->  bf16 [N][K] row-major (coalesced reads)
__global__ __launch_bounds__(256) void cvt_transpose(const float* __restrict__ in,
                                                     __bf16* __restrict__ out,
                                                     int K, int N) {
  int idx = blockIdx.x * blockDim.x + threadIdx.x;
  if (idx >= K * N) return;
  int k = idx / N, n = idx % N;
  out[(size_t)n * K + k] = (__bf16)in[idx];
}

// ---------------------------------------------------------------------------
// qkv = x @ w_qkv ; scatter into Q[B,H,T,D]*1/8 , K[B,H,T,D] , V^T[B,H,D,T] (bf16)
// 32x64 wave tile: 2 M-subtiles x 4 N-subtiles, B fragments reused 2x.
__global__ __launch_bounds__(256) void gemm_qkv(const float* __restrict__ x,
                                                const __bf16* __restrict__ wT,
                                                __bf16* __restrict__ q,
                                                __bf16* __restrict__ kk,
                                                __bf16* __restrict__ vT) {
  const int wave = blockIdx.x * 8 + (threadIdx.x >> 5);
  const int lane = threadIdx.x & 31;
  const int hv = lane >> 4, ln = lane & 15;
  const int NT = (3 * cC) / 64;                 // 48 n-groups
  const int mt = wave / NT, ng = wave % NT;
  const int mbase = mt * 32, nbase = ng * 64;

  const float* arow0 = x + (size_t)(mbase + ln) * cC;
  const float* arow1 = x + (size_t)(mbase + 16 + ln) * cC;
  v8f acc[2][4];
#pragma unroll
  for (int s = 0; s < 2; ++s)
#pragma unroll
    for (int f = 0; f < 4; ++f) acc[s][f] = zero_v8f();

  for (int k0 = 0; k0 < cC; k0 += 32) {
    v16bf a0 = load_a_f32(arow0, k0, hv);
    v16bf a1 = load_a_f32(arow1, k0, hv);
#pragma unroll
    for (int f = 0; f < 4; ++f) {
      const __bf16* brow = wT + (size_t)(nbase + f * 16 + ln) * cC;
      v16bf b = load_b_bf16(brow, k0, hv);
      acc[0][f] = wmma_bf16(a0, b, acc[0][f]);
      acc[1][f] = wmma_bf16(a1, b, acc[1][f]);
    }
  }

  const int sec = nbase / cC;                   // 0=q 1=k 2=v
  const int h = (nbase % cC) / cD;              // tile never crosses heads
#pragma unroll
  for (int s = 0; s < 2; ++s) {
#pragma unroll
    for (int f = 0; f < 4; ++f) {
      const int d = f * 16 + ln;                // nbase % 64 == 0
#pragma unroll
      for (int r = 0; r < 8; ++r) {
        const int m = mbase + s * 16 + r + 8 * hv;
        const int b_ = m / cT, t = m % cT;
        const float val = acc[s][f][r];
        const size_t irow = ((size_t)(b_ * cH + h) * cT + t) * cD + d;
        if (sec == 0)
          q[irow] = (__bf16)(val * 0.125f);     // 1/sqrt(64), exact in bf16
        else if (sec == 1)
          kk[irow] = (__bf16)val;
        else
          vT[((size_t)(b_ * cH + h) * cD + d) * cT + t] = (__bf16)val;
      }
    }
  }
}

// ---------------------------------------------------------------------------
// Flash attention: one wave per 16-query tile, 32 keys per iteration.
__global__ __launch_bounds__(256) void attn_fwd(const __bf16* __restrict__ q,
                                                const __bf16* __restrict__ kmat,
                                                const __bf16* __restrict__ vT,
                                                __bf16* __restrict__ y) {
  __shared__ __bf16 plds[8 * 16 * 32];          // 1KB P-tile per wave
  const int wl = threadIdx.x >> 5;
  const int lane = threadIdx.x & 31;
  const int hv = lane >> 4, ln = lane & 15;
  const int gw = blockIdx.x * 8 + wl;
  const int qt = gw & (cT / 16 - 1);            // 0..127
  const int bh = gw >> 7;                       // 0..B*H-1
  const int qbase = qt * 16;

  __bf16* P = plds + wl * (16 * 32);

  const __bf16* qrow = q + ((size_t)bh * cT + qbase + ln) * cD;
  const v16bf aQ0 = load_a_bf16(qrow, 0, hv);   // d 0..31
  const v16bf aQ1 = load_a_bf16(qrow, 32, hv);  // d 32..63

  float mrow[8], lsum[8];
  v8f o[4];
#pragma unroll
  for (int r = 0; r < 8; ++r) { mrow[r] = -3.0e38f; lsum[r] = 0.f; }
#pragma unroll
  for (int f = 0; f < 4; ++f) o[f] = zero_v8f();

  for (int kb = 0; kb < qbase + 16; kb += 32) {
    const __bf16* krow0 = kmat + ((size_t)bh * cT + kb + ln) * cD;
    const __bf16* krow1 = krow0 + 16 * cD;
    v8f s0 = zero_v8f(), s1 = zero_v8f();
    s0 = wmma_bf16(aQ0, load_b_bf16(krow0, 0, hv), s0);
    s0 = wmma_bf16(aQ1, load_b_bf16(krow0, 32, hv), s0);
    s1 = wmma_bf16(aQ0, load_b_bf16(krow1, 0, hv), s1);
    s1 = wmma_bf16(aQ1, load_b_bf16(krow1, 32, hv), s1);

    // causal mask: C-layout lane = key column, VGPR r -> query row r+8*half
#pragma unroll
    for (int r = 0; r < 8; ++r) {
      const int qg = qbase + r + 8 * hv;
      if (kb + ln > qg)      s0[r] = -3.0e38f;
      if (kb + 16 + ln > qg) s1[r] = -3.0e38f;
    }

    float rm[8];
#pragma unroll
    for (int r = 0; r < 8; ++r) rm[r] = fmaxf(s0[r], s1[r]);
#pragma unroll
    for (int off = 1; off < 16; off <<= 1) {
#pragma unroll
      for (int r = 0; r < 8; ++r) rm[r] = fmaxf(rm[r], __shfl_xor(rm[r], off, 32));
    }

    float scl[8];
#pragma unroll
    for (int r = 0; r < 8; ++r) {
      const float mn = fmaxf(mrow[r], rm[r]);
      scl[r] = __expf(mrow[r] - mn);
      mrow[r] = mn;
    }
#pragma unroll
    for (int r = 0; r < 8; ++r) {
      s0[r] = __expf(s0[r] - mrow[r]);
      s1[r] = __expf(s1[r] - mrow[r]);
    }
    float rs[8];
#pragma unroll
    for (int r = 0; r < 8; ++r) rs[r] = s0[r] + s1[r];
#pragma unroll
    for (int off = 1; off < 16; off <<= 1) {
#pragma unroll
      for (int r = 0; r < 8; ++r) rs[r] += __shfl_xor(rs[r], off, 32);
    }
#pragma unroll
    for (int r = 0; r < 8; ++r) lsum[r] = lsum[r] * scl[r] + rs[r];
#pragma unroll
    for (int f = 0; f < 4; ++f)
#pragma unroll
      for (int r = 0; r < 8; ++r) o[f][r] *= scl[r];

    // C-fragment -> A-fragment re-layout via per-wave LDS tile (DS in-order)
#pragma unroll
    for (int r = 0; r < 8; ++r) {
      P[(r + 8 * hv) * 32 + ln]      = (__bf16)s0[r];
      P[(r + 8 * hv) * 32 + 16 + ln] = (__bf16)s1[r];
    }
    const v16bf aP = load_a_bf16(P + ln * 32, 0, hv);

#pragma unroll
    for (int f = 0; f < 4; ++f) {
      const __bf16* vrow = vT + ((size_t)bh * cD + f * 16 + ln) * cT;
      o[f] = wmma_bf16(aP, load_b_bf16(vrow, kb, hv), o[f]);
    }
  }

  const int b_ = bh >> 4, h = bh & (cH - 1);
#pragma unroll
  for (int r = 0; r < 8; ++r) {
    const float inv = 1.0f / lsum[r];
    const int t = qbase + r + 8 * hv;
#pragma unroll
    for (int f = 0; f < 4; ++f) {
      y[(size_t)(b_ * cT + t) * cC + h * cD + f * 16 + ln] = (__bf16)(o[f][r] * inv);
    }
  }
}

// ---------------------------------------------------------------------------
// out = y @ w_out + b_out (fp32 output); 32x64 wave tile, B reused 2x.
__global__ __launch_bounds__(256) void gemm_out(const __bf16* __restrict__ y,
                                                const __bf16* __restrict__ wT,
                                                const float* __restrict__ bias,
                                                float* __restrict__ out) {
  const int wave = blockIdx.x * 8 + (threadIdx.x >> 5);
  const int lane = threadIdx.x & 31;
  const int hv = lane >> 4, ln = lane & 15;
  const int NT = cC / 64;                       // 16 n-groups
  const int mt = wave / NT, ng = wave % NT;
  const int mbase = mt * 32, nbase = ng * 64;

  const __bf16* arow0 = y + (size_t)(mbase + ln) * cC;
  const __bf16* arow1 = y + (size_t)(mbase + 16 + ln) * cC;
  v8f acc[2][4];
#pragma unroll
  for (int s = 0; s < 2; ++s)
#pragma unroll
    for (int f = 0; f < 4; ++f) acc[s][f] = zero_v8f();

  for (int k0 = 0; k0 < cC; k0 += 32) {
    v16bf a0 = load_a_bf16(arow0, k0, hv);
    v16bf a1 = load_a_bf16(arow1, k0, hv);
#pragma unroll
    for (int f = 0; f < 4; ++f) {
      const __bf16* brow = wT + (size_t)(nbase + f * 16 + ln) * cC;
      v16bf b = load_b_bf16(brow, k0, hv);
      acc[0][f] = wmma_bf16(a0, b, acc[0][f]);
      acc[1][f] = wmma_bf16(a1, b, acc[1][f]);
    }
  }
#pragma unroll
  for (int s = 0; s < 2; ++s) {
#pragma unroll
    for (int f = 0; f < 4; ++f) {
      const int n = nbase + f * 16 + ln;
      const float bs = bias[n];
#pragma unroll
      for (int r = 0; r < 8; ++r) {
        out[(size_t)(mbase + s * 16 + r + 8 * hv) * cC + n] = acc[s][f][r] + bs;
      }
    }
  }
}

// ---------------------------------------------------------------------------
extern "C" void kernel_launch(void* const* d_in, const int* in_sizes, int n_in,
                              void* d_out, int out_size, void* d_ws, size_t ws_size,
                              hipStream_t stream) {
  const float* x     = (const float*)d_in[0];
  const float* w_qkv = (const float*)d_in[1];
  const float* w_out = (const float*)d_in[2];
  const float* b_out = (const float*)d_in[3];
  float* out = (float*)d_out;

  char* ws = (char*)d_ws;
  const size_t SZ_WQKVT = (size_t)3 * cC * cC * 2;        // 6 MB
  const size_t SZ_WOUTT = (size_t)cC * cC * 2;            // 2 MB
  const size_t SZ_HEAD  = (size_t)cB * cH * cT * cD * 2;  // 8 MB each
  __bf16* wqkvT = (__bf16*)(ws);
  __bf16* woutT = (__bf16*)(ws + SZ_WQKVT);
  __bf16* qb    = (__bf16*)(ws + SZ_WQKVT + SZ_WOUTT);
  __bf16* kb    = (__bf16*)(ws + SZ_WQKVT + SZ_WOUTT + SZ_HEAD);
  __bf16* vTb   = (__bf16*)(ws + SZ_WQKVT + SZ_WOUTT + 2 * SZ_HEAD);
  __bf16* yb    = (__bf16*)(ws + SZ_WQKVT + SZ_WOUTT + 3 * SZ_HEAD);

  {
    int n = cC * 3 * cC;
    cvt_transpose<<<(n + 255) / 256, 256, 0, stream>>>(w_qkv, wqkvT, cC, 3 * cC);
  }
  {
    int n = cC * cC;
    cvt_transpose<<<(n + 255) / 256, 256, 0, stream>>>(w_out, woutT, cC, cC);
  }
  {
    const int waves = (cB * cT / 32) * ((3 * cC) / 64);   // 6144 waves
    gemm_qkv<<<waves / 8, 256, 0, stream>>>(x, wqkvT, qb, kb, vTb);
  }
  {
    const int waves = cB * cH * (cT / 16);                // 4096 waves
    attn_fwd<<<waves / 8, 256, 0, stream>>>(qb, kb, vTb, yb);
  }
  {
    const int waves = (cB * cT / 32) * (cC / 64);         // 2048 waves
    gemm_out<<<waves / 8, 256, 0, stream>>>(yb, woutT, b_out, out);
  }
}
